// VGG_SFCN_89163521065652
// MI455X (gfx1250) — matrix operand order repaired and checked by reading, not compiled, not run on hardware
//
#include <hip/hip_runtime.h>

// ---------------------------------------------------------------------------
// VGG-SFCN for MI455X (gfx1250).
//   * 3x3 convs (Cin % 32 == 0) as implicit GEMM on v_wmma_f32_16x16x32_f16.
//     Weights are pre-packed per layer into f16 B-fragment order so the inner
//     loop is 2x ds_load_b128 (A) + 2x global_load_b128 (B) + v_wmma.
//   * First conv (3ch), maxpools: scalar kernels (negligible FLOPs).
//   * convDU / convLR bidirectional scans: sequential per-slice kernels.
// ---------------------------------------------------------------------------

typedef __attribute__((ext_vector_type(16))) _Float16 v16h;
typedef __attribute__((ext_vector_type(8)))  float    v8f;

// ---------------------------------------------------------------------------
// Repack fp32 OIHW weights -> f16 in the B-fragment order conv3x3_wmma uses:
//   pw[((((co>>4)*(C/32) + ci0/32)*9 + tap)*32 + lane)*16 + j]
// where lane: n = lane&15, half = lane>>4; element j holds K = 16*half + j,
// i.e. input channel ci0 + 16*half + j, out channel co_base + n.
// ---------------------------------------------------------------------------
__global__ void pack_wgt_f16(const float* __restrict__ wgt, _Float16* __restrict__ pw,
                             int C, int Co) {
  size_t total = (size_t)Co * C * 9;
  size_t idx = (size_t)blockIdx.x * blockDim.x + threadIdx.x;
  if (idx >= total) return;
  int j     = (int)(idx & 15);
  int lane  = (int)((idx >> 4) & 31);
  int t     = (int)((idx >> 9) % 9);
  size_t r  = (idx >> 9) / 9;
  int cb    = (int)(r % (size_t)(C >> 5));
  int ctile = (int)(r / (size_t)(C >> 5));
  int nn = lane & 15, hf = lane >> 4;
  int co = ctile * 16 + nn;
  int ci = cb * 32 + 16 * hf + j;
  pw[idx] = (_Float16)wgt[((size_t)co * C + ci) * 9 + t];
}

// ---------------------------------------------------------------------------
// WMMA conv3x3 (pad = dil, 'same'), fused bias + ReLU.
// Block = 256 threads = 8 waves: 2 pixel tiles (16 px each) x 4 co tiles (16).
// Grid: (W/32, H, 2 * Co/64).  Requires C % 32 == 0, Co % 64 == 0, W % 32 == 0.
// ---------------------------------------------------------------------------
__launch_bounds__(256)
__global__ void conv3x3_wmma(const float* __restrict__ in, float* __restrict__ out,
                             const _Float16* __restrict__ pwgt,
                             const float* __restrict__ bias,
                             int C, int Co, int H, int W, int dil) {
  // [pixel-tile][row(3)][col(<=20)][ch(32)] f16, ch contiguous -> b128 A loads
  __shared__ _Float16 lin[2][3][20][32];

  const int tid  = threadIdx.x;
  const int wave = tid >> 5;
  const int lane = tid & 31;
  const int hf   = lane >> 4;   // lane half (WMMA layout selector)
  const int mn   = lane & 15;   // M (pixel) for A, N (out-ch) for B/C/D
  const int pt   = wave & 1;    // pixel tile
  const int ct   = wave >> 1;   // channel tile

  const int w0      = blockIdx.x * 32;
  const int h       = blockIdx.y;
  const int coTiles = Co >> 6;
  const int n       = blockIdx.z / coTiles;
  const int cog     = blockIdx.z % coTiles;

  const int wp    = w0 + 16 * pt;           // wave's pixel base
  const int co_w  = (cog << 6) + (ct << 4); // wave's out-channel base
  const int cw    = 16 + 2 * dil;           // staged cols per pixel tile
  const int per_t = 3 * cw * 32;            // staged halfwords per tile

  const v16h* pw = (const v16h*)pwgt;       // one v16h = one lane's B fragment

  v8f acc = {};

  for (int ci0 = 0; ci0 < C; ci0 += 32) {
    __syncthreads();
    // Cooperative stage: rows h-d,h,h+d; cols [tile_base-d, tile_base+15+d]; 32 ch.
    for (int e = tid; e < 2 * per_t; e += 256) {
      int p   = e / per_t;
      int r2  = e % per_t;
      int ch  = r2 / (3 * cw);
      int rem = r2 % (3 * cw);
      int r   = rem / cw;
      int col = rem % cw;
      int gh  = h + (r - 1) * dil;
      int gw  = w0 + 16 * p - dil + col;
      float val = 0.f;
      if ((unsigned)gh < (unsigned)H && (unsigned)gw < (unsigned)W)
        val = in[(((size_t)n * C + ci0 + ch) * H + gh) * W + gw];
      lin[p][r][col][ch] = (_Float16)val;
    }
    __syncthreads();

    // Warm next channel slab in L2 while we compute this one.
    if (ci0 + 32 < C) {
      const float* nxt =
          &in[(((size_t)n * C + ci0 + 32 + (tid >> 3)) * H + h) * W + w0];
      __builtin_prefetch(nxt, 0, 1);
    }

    // B fragment base for this (co tile, ci block): 9 taps contiguous.
    const size_t bbase =
        ((size_t)(co_w >> 4) * (C >> 5) + (ci0 >> 5)) * 9 * 32 + lane;

    // 9 taps: each tap is one K=32 WMMA slab (32 input channels).
    for (int t = 0; t < 9; ++t) {
      const int th = t / 3, tw = t % 3;
      v16h a;
#pragma unroll
      for (int v = 0; v < 8; ++v) {
        // A (16x32 f16): lane<16 -> K 0..15; lane>=16 -> K 16..31
        int ka = 2 * v + 16 * hf;
        a[2 * v]     = lin[pt][th][mn + tw * dil][ka];
        a[2 * v + 1] = lin[pt][th][mn + tw * dil][ka + 1];
      }
      v16h b = pw[bbase + (size_t)t * 32];   // 32B aligned, 2x global_load_b128
      acc = __builtin_amdgcn_wmma_f32_16x16x32_f16(
          false, a, false, b, (short)0, acc, false, false);
    }
  }

  // C/D layout: VGPR r holds M = r + 8*half, N = mn. Fused bias + ReLU.
  float bi = bias[co_w + mn];
#pragma unroll
  for (int r = 0; r < 8; ++r) {
    int m = r + 8 * hf;
    float v = acc[r] + bi;
    out[(((size_t)n * Co + co_w + mn) * H + h) * W + wp + m] = v > 0.f ? v : 0.f;
  }
}

// ---------------------------------------------------------------------------
// First conv: 3 -> 64 channels (K=27, not WMMA-friendly; 0.4% of total FLOPs).
// ---------------------------------------------------------------------------
__global__ void conv3x3_c3(const float* __restrict__ in, float* __restrict__ out,
                           const float* __restrict__ wgt, const float* __restrict__ bias,
                           int H, int W) {
  size_t idx = (size_t)blockIdx.x * blockDim.x + threadIdx.x;
  size_t HW = (size_t)H * W;
  size_t total = 2ull * 64 * HW;
  if (idx >= total) return;
  int w  = (int)(idx % W);
  int h  = (int)((idx / W) % H);
  int co = (int)((idx / HW) % 64);
  int n  = (int)(idx / (HW * 64));
  float acc = bias[co];
  for (int ci = 0; ci < 3; ++ci)
    for (int kh = 0; kh < 3; ++kh) {
      int gh = h + kh - 1;
      if ((unsigned)gh >= (unsigned)H) continue;
      for (int kw = 0; kw < 3; ++kw) {
        int gw = w + kw - 1;
        if ((unsigned)gw >= (unsigned)W) continue;
        acc += wgt[((co * 3 + ci) * 3 + kh) * 3 + kw] *
               in[(((size_t)n * 3 + ci) * H + gh) * W + gw];
      }
    }
  out[idx] = fmaxf(acc, 0.f);
}

// ---------------------------------------------------------------------------
// 2x2 max pool, stride 2.
// ---------------------------------------------------------------------------
__global__ void maxpool2(const float* __restrict__ in, float* __restrict__ out,
                         int C, int H, int W) {
  int Ho = H >> 1, Wo = W >> 1;
  size_t total = 2ull * C * Ho * Wo;
  size_t idx = (size_t)blockIdx.x * blockDim.x + threadIdx.x;
  if (idx >= total) return;
  int wo = (int)(idx % Wo);
  int ho = (int)((idx / Wo) % Ho);
  int c  = (int)((idx / ((size_t)Wo * Ho)) % C);
  int n  = (int)(idx / ((size_t)Wo * Ho * C));
  const float* p = &in[(((size_t)n * C + c) * H + 2 * ho) * W + 2 * wo];
  out[idx] = fmaxf(fmaxf(p[0], p[1]), fmaxf(p[W], p[W + 1]));
}

// ---------------------------------------------------------------------------
// Scan support. Slices are [n=2][c=64][j=64]. Addressing modes:
//   0: NCHW slice over H (convDU; j = w)
//   1: NCHW slice over W (convLR; j = h)
//   2: contiguous F storage: i*8192 + (n*64+c)*64 + j
// ---------------------------------------------------------------------------
__device__ __forceinline__ size_t slice_idx(int mode, int i, int n, int c, int j) {
  if (mode == 0) return ((size_t)(n * 64 + c) * 64 + i) * 64 + j;
  if (mode == 1) return ((size_t)(n * 64 + c) * 64 + j) * 64 + i;
  return (size_t)i * 8192 + (size_t)(n * 64 + c) * 64 + j;
}

__global__ void copy_slice(const float* __restrict__ src, int smode, int si,
                           float* __restrict__ dst, int dmode, int di) {
  int e = blockIdx.x * 256 + threadIdx.x;   // 8192 elements
  if (e >= 8192) return;
  int n = e >> 12, c = (e >> 6) & 63, j = e & 63;
  dst[slice_idx(dmode, di, n, c, j)] = src[slice_idx(smode, si, n, c, j)];
}

// One scan step: dst = relu(conv1d_k9(prev) + bias) + src.
// Grid = 8 blocks: block b -> n = b/4, co quarter q = b%4 (16 co).
__launch_bounds__(256)
__global__ void scan_step(const float* __restrict__ prev, int pmode, int pi,
                          const float* __restrict__ src,  int smode, int si,
                          float* __restrict__ dst,        int dmode, int di,
                          const float* __restrict__ w, const float* __restrict__ bias) {
  __shared__ float sprev[64 * 64];     // carry slice for this n (16 KB)
  __shared__ float sw[16 * 64 * 9];    // weight block for 16 out-ch (36 KB)
  const int tid = threadIdx.x;
  const int n = blockIdx.x >> 2;
  const int q = blockIdx.x & 3;

  for (int e = tid; e < 4096; e += 256) {
    int c = e >> 6, j = e & 63;
    sprev[e] = prev[slice_idx(pmode, pi, n, c, j)];
  }
  for (int e = tid; e < 16 * 576; e += 256)
    sw[e] = w[(size_t)(q * 16) * 576 + e];
  __syncthreads();

  for (int o = tid; o < 1024; o += 256) {
    int co = q * 16 + (o >> 6);
    int j  = o & 63;
    float acc = bias[co];
    const float* wc = &sw[(co - q * 16) * 576];
    for (int ci = 0; ci < 64; ++ci) {
      const float* pr = &sprev[ci * 64];
      const float* wk = &wc[ci * 9];
#pragma unroll
      for (int k = 0; k < 9; ++k) {
        int jj = j + k - 4;
        if (jj >= 0 && jj < 64) acc += wk[k] * pr[jj];
      }
    }
    float v = fmaxf(acc, 0.f) + src[slice_idx(smode, si, n, co, j)];
    dst[slice_idx(dmode, di, n, co, j)] = v;
  }
}

// ---------------------------------------------------------------------------
// Host orchestration
// ---------------------------------------------------------------------------
static const int FIN[10]  = {3, 64, 64, 128, 128, 256, 256, 256, 512, 512};
static const int FOUT[10] = {64, 64, 128, 128, 256, 256, 256, 512, 512, 512};
static const int BIN[6]   = {512, 512, 512, 512, 256, 128};
static const int BOUT[6]  = {512, 512, 512, 256, 128, 64};

static inline void launch_conv(const float* in, float* out, const float* w,
                               _Float16* pwbuf, const float* b,
                               int C, int Co, int H, int W, int dil,
                               hipStream_t s) {
  size_t wtot = (size_t)Co * C * 9;
  pack_wgt_f16<<<dim3((unsigned)((wtot + 255) / 256)), 256, 0, s>>>(w, pwbuf, C, Co);
  dim3 g(W / 32, H, 2 * (Co / 64));
  conv3x3_wmma<<<g, 256, 0, s>>>(in, out, pwbuf, b, C, Co, H, W, dil);
}

static inline void launch_pool(const float* in, float* out, int C, int H, int W,
                               hipStream_t s) {
  size_t total = 2ull * C * (H / 2) * (W / 2);
  maxpool2<<<dim3((unsigned)((total + 255) / 256)), 256, 0, s>>>(in, out, C, H, W);
}

extern "C" void kernel_launch(void* const* d_in, const int* in_sizes, int n_in,
                              void* d_out, int out_size, void* d_ws, size_t ws_size,
                              hipStream_t stream) {
  (void)in_sizes; (void)out_size; (void)ws_size;
  const float* x = (const float*)d_in[0];

  const float *fw[10], *fb[10], *bw[6], *bb[6], *du_w, *du_b, *lr_w, *lr_b;
  if (n_in >= 37) {  // pytree-flattened tuples
    for (int i = 0; i < 10; ++i) { fw[i] = (const float*)d_in[1 + i];
                                   fb[i] = (const float*)d_in[11 + i]; }
    for (int i = 0; i < 6;  ++i) { bw[i] = (const float*)d_in[21 + i];
                                   bb[i] = (const float*)d_in[27 + i]; }
    du_w = (const float*)d_in[33]; du_b = (const float*)d_in[34];
    lr_w = (const float*)d_in[35]; lr_b = (const float*)d_in[36];
  } else {           // one concatenated buffer per dict entry
    const float* p = (const float*)d_in[1];
    for (int i = 0; i < 10; ++i) { fw[i] = p; p += (size_t)FOUT[i] * FIN[i] * 9; }
    p = (const float*)d_in[2];
    for (int i = 0; i < 10; ++i) { fb[i] = p; p += FOUT[i]; }
    p = (const float*)d_in[3];
    for (int i = 0; i < 6;  ++i) { bw[i] = p; p += (size_t)BOUT[i] * BIN[i] * 9; }
    p = (const float*)d_in[4];
    for (int i = 0; i < 6;  ++i) { bb[i] = p; p += BOUT[i]; }
    du_w = (const float*)d_in[5]; du_b = (const float*)d_in[6];
    lr_w = (const float*)d_in[7]; lr_b = (const float*)d_in[8];
  }

  const size_t SZ   = 2ull * 64 * 512 * 512 * sizeof(float);  // 134 MB ping buffer
  const size_t F_SZ = 64ull * 8192 * sizeof(float);           // 2 MB scan slices
  float*    A0 = (float*)d_ws;
  float*    A1 = (float*)((char*)d_ws + SZ);
  float*    F  = (float*)((char*)d_ws + 2 * SZ);
  _Float16* PW = (_Float16*)((char*)d_ws + 2 * SZ + F_SZ);    // <= 4.5 MB packed w
  float*    O  = (float*)d_out;

  // ---- Front VGG (dil=1) ----
  {
    size_t total = 2ull * 64 * 512 * 512;
    conv3x3_c3<<<dim3((unsigned)((total + 255) / 256)), 256, 0, stream>>>(
        x, A0, fw[0], fb[0], 512, 512);
  }
  launch_conv(A0, A1, fw[1], PW, fb[1], 64, 64, 512, 512, 1, stream);
  launch_pool(A1, A0, 64, 512, 512, stream);                        // -> 256
  launch_conv(A0, A1, fw[2], PW, fb[2], 64, 128, 256, 256, 1, stream);
  launch_conv(A1, A0, fw[3], PW, fb[3], 128, 128, 256, 256, 1, stream);
  launch_pool(A0, A1, 128, 256, 256, stream);                       // -> 128
  launch_conv(A1, A0, fw[4], PW, fb[4], 128, 256, 128, 128, 1, stream);
  launch_conv(A0, A1, fw[5], PW, fb[5], 256, 256, 128, 128, 1, stream);
  launch_conv(A1, A0, fw[6], PW, fb[6], 256, 256, 128, 128, 1, stream);
  launch_pool(A0, A1, 256, 128, 128, stream);                       // -> 64
  launch_conv(A1, A0, fw[7], PW, fb[7], 256, 512, 64, 64, 1, stream);
  launch_conv(A0, A1, fw[8], PW, fb[8], 512, 512, 64, 64, 1, stream);
  launch_conv(A1, A0, fw[9], PW, fb[9], 512, 512, 64, 64, 1, stream);

  // ---- Back VGG (dil=2) ----
  launch_conv(A0, A1, bw[0], PW, bb[0], 512, 512, 64, 64, 2, stream);
  launch_conv(A1, A0, bw[1], PW, bb[1], 512, 512, 64, 64, 2, stream);
  launch_conv(A0, A1, bw[2], PW, bb[2], 512, 512, 64, 64, 2, stream);
  launch_conv(A1, A0, bw[3], PW, bb[3], 512, 256, 64, 64, 2, stream);
  launch_conv(A0, A1, bw[4], PW, bb[4], 256, 128, 64, 64, 2, stream);
  launch_conv(A1, A0, bw[5], PW, bb[5], 128, 64, 64, 64, 2, stream);
  // activations (2,64,64,64) now in A0

  // ---- convDU: scan over H, conv along W (mode 0), A0 -> A1 ----
  copy_slice<<<32, 256, 0, stream>>>(A0, 0, 0, F, 2, 0);
  for (int i = 1; i < 64; ++i)
    scan_step<<<8, 256, 0, stream>>>(F, 2, i - 1, A0, 0, i, F, 2, i, du_w, du_b);
  copy_slice<<<32, 256, 0, stream>>>(F, 2, 63, A1, 0, 63);
  for (int i = 62; i >= 0; --i)
    scan_step<<<8, 256, 0, stream>>>(A1, 0, i + 1, F, 2, i, A1, 0, i, du_w, du_b);

  // ---- convLR: scan over W, conv along H (mode 1), A1 -> d_out ----
  copy_slice<<<32, 256, 0, stream>>>(A1, 1, 0, F, 2, 0);
  for (int i = 1; i < 64; ++i)
    scan_step<<<8, 256, 0, stream>>>(F, 2, i - 1, A1, 1, i, F, 2, i, lr_w, lr_b);
  copy_slice<<<32, 256, 0, stream>>>(F, 2, 63, O, 1, 63);
  for (int i = 62; i >= 0; --i)
    scan_step<<<8, 256, 0, stream>>>(O, 1, i + 1, F, 2, i, O, 1, i, lr_w, lr_b);
}
